// RNN_17179869511
// MI455X (gfx1250) — matrix operand order repaired and checked by reading, not compiled, not run on hardware
//
#include <hip/hip_runtime.h>

#define B_ 4096
#define T_ 512
#define I_ 28
#define H_ 64
#define C_ 10
#define G_ 256   // 4*H (gate outputs, PyTorch order i,f,g,o)
#define KP_ 96   // padded K: 32 (x, pad 28->32) + 64 (h)

typedef __attribute__((ext_vector_type(16))) __bf16 v16bf;
typedef __attribute__((ext_vector_type(8)))  __bf16 v8bf;
typedef __attribute__((ext_vector_type(8)))  float  v8f;
typedef __attribute__((ext_vector_type(4)))  float  f32x4;

#if __has_builtin(__builtin_amdgcn_tanhf)
__device__ __forceinline__ float tanh_(float x) { return __builtin_amdgcn_tanhf(x); }
#else
__device__ __forceinline__ float tanh_(float x) {
    // tanh(x) = 2/(1+exp(-2x)) - 1,  exp via v_exp_f32 (exp2)
    return 2.0f * __builtin_amdgcn_rcpf(1.0f + __builtin_amdgcn_exp2f(-2.8853900817779268f * x)) - 1.0f;
}
#endif
__device__ __forceinline__ float sigm_(float x) { return __builtin_fmaf(0.5f, tanh_(0.5f * x), 0.5f); }

// One block (4 waves) owns a 16-row batch tile for all T steps.
// Wave w owns N-tiles {w, w+4, w+8, w+12} == the i/f/g/o gates of hidden tile w,
// so the gate nonlinearity needs NO cross-wave exchange. Only h is shared,
// via a double-buffered LDS tile with one barrier per timestep.
// Per step per wave: 3 K-chunks x 4 N-tiles = 12 v_wmma_f32_16x16x32_bf16.
__global__ __launch_bounds__(128)
void lstm_fused_wmma(const float* __restrict__ x,
                     const float* __restrict__ W_ih,
                     const float* __restrict__ W_hh,
                     const float* __restrict__ b_ih,
                     const float* __restrict__ b_hh,
                     const float* __restrict__ W_out,
                     const float* __restrict__ b_out,
                     float* __restrict__ out)
{
    __shared__ __align__(64) __bf16 sW[G_ * KP_];   // Wcomb^T: sW[n][k], 48 KB bf16
    __shared__ float  sBias[G_];                    // b_ih + b_hh
    __shared__ __align__(64) __bf16 sH[2][16 * H_]; // double-buffered h tile (bf16)

    const int tid  = threadIdx.x;
    const int lane = tid & 31;
    const int wave = tid >> 5;         // 0..3 : hidden tile / strided gate owner
    const int lo16 = lane & 15;
    const int hi   = lane >> 4;        // 0: lanes 0-15, 1: lanes 16-31

    // ---- stage combined weights + bias into LDS (bf16) ----
    for (int n = tid; n < G_; n += 128) {
        #pragma unroll
        for (int k = 0; k < 32; ++k)
            sW[n*KP_ + k] = (k < I_) ? (__bf16)W_ih[n*I_ + k] : (__bf16)0.0f;
        #pragma unroll
        for (int k = 0; k < H_; ++k)
            sW[n*KP_ + 32 + k] = (__bf16)W_hh[n*H_ + k];
        sBias[n] = b_ih[n] + b_hh[n];
    }
    {
        __bf16* z = &sH[0][0];
        for (int idx = tid; idx < 2*16*H_; idx += 128) z[idx] = (__bf16)0.0f;
    }
    __syncthreads();

    // ---- preload this wave's loop-invariant B fragments (12 x v16bf = 96 VGPRs) ----
    // B fragment, contiguous K order (as A): lane holds N = lane&15, 16 consecutive K values
    // starting at 16*(lane>>4) within the 32-K chunk -> one 32B LDS read.
    v16bf Bf[4][3];
    float bj[4];
    #pragma unroll
    for (int d = 0; d < 4; ++d) {          // d: gate (0=i,1=f,2=g,3=o), tile j = wave + 4d
        const int n = 16*(wave + 4*d) + lo16;
        #pragma unroll
        for (int q = 0; q < 3; ++q) {
            const int kb = 32*q + 16*hi;
            Bf[d][q] = *(const v16bf*)&sW[n*KP_ + kb];
        }
        bj[d] = sBias[n];
    }

    const int  m    = lo16;                          // A/D row id within tile
    const long b    = (long)blockIdx.x * 16 + m;     // batch row
    const float* xrow = x + (size_t)b * T_ * I_;
    const int  kb0  = hi ? 16 : 0;                   // A chunk0 K-group base

    float cst[8];                                    // cell state, hidden tile = wave
    #pragma unroll
    for (int r = 0; r < 8; ++r) cst[r] = 0.0f;

    for (int t = 0; t < T_; ++t) {
        const float* xr = xrow + (size_t)t * I_;
        const __bf16* sHr = &sH[t & 1][0];           // h_{t-1}
        __bf16*       sHw2 = &sH[(t + 1) & 1][0];    // h_t

        // A chunk0 (x slice): lane needs K {kb0..kb0+15}; K>=28 -> 0
        f32x4 l0 = *(const f32x4*)(xr + kb0);
        f32x4 l1 = *(const f32x4*)(xr + kb0 + 4);
        f32x4 l2 = *(const f32x4*)(xr + kb0 + 8);
        f32x4 l3 = {0.f, 0.f, 0.f, 0.f};
        if (!hi) l3 = *(const f32x4*)(xr + 12);
        if (t + 1 < T_) __builtin_prefetch(xr + I_, 0, 0);   // global_prefetch_b8

        v16bf a0;
        #pragma unroll
        for (int r = 0; r < 4; ++r) {
            a0[r]    = (__bf16)l0[r];
            a0[4+r]  = (__bf16)l1[r];
            a0[8+r]  = (__bf16)l2[r];
            a0[12+r] = (__bf16)l3[r];
        }

        // A chunks 1,2 (h slice) from shared LDS tile, contiguous K (16*hi + s), as B
        v16bf a1, a2;
        {
            const int hb1 = 16*hi;
            v8bf g0 = *(const v8bf*)&sHr[m*H_ + hb1];
            v8bf g1 = *(const v8bf*)&sHr[m*H_ + hb1 + 8];
            const int hb2 = 32 + 16*hi;
            v8bf g2 = *(const v8bf*)&sHr[m*H_ + hb2];
            v8bf g3 = *(const v8bf*)&sHr[m*H_ + hb2 + 8];
            #pragma unroll
            for (int r = 0; r < 8; ++r) {
                a1[r] = g0[r]; a1[8+r] = g1[r];
                a2[r] = g2[r]; a2[8+r] = g3[r];
            }
        }

        // 12 WMMAs: 4 gate tiles x 3 K-chunks, bias-seeded accumulators
        v8f acc[4];
        #pragma unroll
        for (int d = 0; d < 4; ++d) {
            v8f c;
            #pragma unroll
            for (int r = 0; r < 8; ++r) c[r] = bj[d];
            c = __builtin_amdgcn_wmma_f32_16x16x32_bf16(false, a0, false, Bf[d][0], (short)0, c, false, false);
            c = __builtin_amdgcn_wmma_f32_16x16x32_bf16(false, a1, false, Bf[d][1], (short)0, c, false, false);
            c = __builtin_amdgcn_wmma_f32_16x16x32_bf16(false, a2, false, Bf[d][2], (short)0, c, false, false);
            acc[d] = c;
        }

        // gate activations + cell update for hidden tile k = wave; write h_t
        #pragma unroll
        for (int r = 0; r < 8; ++r) {
            float ig = sigm_(acc[0][r]);
            float fg = sigm_(acc[1][r]);
            float gg = tanh_(acc[2][r]);
            float og = sigm_(acc[3][r]);
            float cv = fg * cst[r] + ig * gg;
            cst[r] = cv;
            float hv = og * tanh_(cv);
            const int mrow = r + 8*hi;               // D layout row
            sHw2[mrow*H_ + 16*wave + lo16] = (__bf16)hv;
        }

        __syncthreads();                             // publish h_t to all 4 waves
    }

    // ---- output projection: out[b,c] = h_T[b,:] . W_out[c,:] + b_out[c] ----
    // h_T lives in sH[T_ & 1] (written as buffer (511+1)&1 == 0 == T_&1).
    if (wave == 0) {
        const __bf16* hT = &sH[T_ & 1][0];
        const int cbase = hi ? 5 : 0;
        #pragma unroll
        for (int ci = 0; ci < 5; ++ci) {
            const int cc = cbase + ci;
            float s = b_out[cc];
            #pragma unroll
            for (int k = 0; k < H_; ++k)
                s += (float)hT[m*H_ + k] * W_out[cc*H_ + k];
            out[b*C_ + cc] = s;
        }
    }
}

extern "C" void kernel_launch(void* const* d_in, const int* in_sizes, int n_in,
                              void* d_out, int out_size, void* d_ws, size_t ws_size,
                              hipStream_t stream) {
    const float* x     = (const float*)d_in[0];
    const float* W_ih  = (const float*)d_in[1];
    const float* W_hh  = (const float*)d_in[2];
    const float* b_ih  = (const float*)d_in[3];
    const float* b_hh  = (const float*)d_in[4];
    const float* W_out = (const float*)d_in[5];
    const float* b_out = (const float*)d_in[6];
    float* out = (float*)d_out;

    // one block per 16-row batch tile: 256 blocks x 4 waves = 1024 waves
    lstm_fused_wmma<<<B_ / 16, 128, 0, stream>>>(x, W_ih, W_hh, b_ih, b_hh, W_out, b_out, out);
}
